// LSTMAtt_Classifier_72370198938142
// MI455X (gfx1250) — matrix operand order repaired and checked by reading, not compiled, not run on hardware
//
#include <hip/hip_runtime.h>
#include <cstdint>
#include <cstddef>

// ---------------------------------------------------------------------------
// Model dims (fixed by the reference)
// ---------------------------------------------------------------------------
constexpr int B  = 8;
constexpr int S  = 128;
constexpr int D  = 512;
constexpr int N  = 64;
constexpr int H  = 128;
constexpr int BS = B * S;          // 1024 rows
constexpr int G4 = 4 * H;          // 512 gate cols

typedef __bf16 bf16;
typedef __attribute__((ext_vector_type(16))) __bf16 bf16x16;
typedef __attribute__((ext_vector_type(8)))  __bf16 bf16x8;
typedef __attribute__((ext_vector_type(4)))  __bf16 bf16x4;
typedef __attribute__((ext_vector_type(8)))  float   f32x8;

#define DEV static __device__ __forceinline__

DEV float sigmoidf_(float x) { return 1.0f / (1.0f + __expf(-x)); }

// ---------------------------------------------------------------------------
// WMMA tile loaders (CDNA5 wave32, v_wmma_f32_16x16x32_bf16 operand layouts)
//
// A (16x32 bf16):  lane = m + 16*h (m = row).  elements e     -> K = 16h+e
// B (32x16 bf16):  lane = c + 16*h (c = col).  elements e     -> K = 16h+e
// C/D (16x16 f32): lane = c + 16*h.            elements e     -> row 8h+e
// ---------------------------------------------------------------------------
DEV bf16x16 load_a_tile(const bf16* __restrict__ base, int ldr, int k0, int lane) {
  int m = lane & 15, h = lane >> 4;
  const bf16* p = base + (size_t)m * ldr + k0 + 16 * h;
  bf16x8 lo = *reinterpret_cast<const bf16x8*>(p);         // K = k0+16h .. +7
  bf16x8 hi = *reinterpret_cast<const bf16x8*>(p + 8);     // K = k0+16h+8 .. +7
  bf16x16 r;
#pragma unroll
  for (int e = 0; e < 8; ++e) { r[e] = lo[e]; r[e + 8] = hi[e]; }
  return r;
}

// B from W stored [Nout, K] row-major (i.e. computing A @ W^T): contiguous K.
DEV bf16x16 load_b_trans(const bf16* __restrict__ W, int ldw, int n0, int k0, int lane) {
  int c = lane & 15, h = lane >> 4;
  const bf16* p = W + (size_t)(n0 + c) * ldw + k0 + 16 * h;
  bf16x8 lo = *reinterpret_cast<const bf16x8*>(p);
  bf16x8 hi = *reinterpret_cast<const bf16x8*>(p + 8);
  bf16x16 r;
#pragma unroll
  for (int e = 0; e < 8; ++e) { r[e] = lo[e]; r[e + 8] = hi[e]; }
  return r;
}

// B from matrix stored [K, N] row-major (non-transposed): strided by ldb.
DEV bf16x16 load_b_norm(const bf16* __restrict__ Bm, int ldb, int n0, int k0, int lane) {
  int c = lane & 15, h = lane >> 4;
  const bf16* p = Bm + (size_t)(k0 + 16 * h) * ldb + n0 + c;
  bf16x16 r;
#pragma unroll
  for (int e = 0; e < 16; ++e) r[e] = p[(size_t)e * ldb];
  return r;
}

// ---------------------------------------------------------------------------
// Generic bf16 WMMA GEMM:  C(f32)[M,N] = A(bf16)[M,K] @ B(^T) + bias1 + bias2
// Each wave computes a 16x64 output panel (4 accumulators) so one A-tile
// register load feeds 4 WMMAs per K-step. 8 waves per block; batch = grid.z.
// M % 16 == 0, N % 64 == 0, K % 32 == 0.
// ---------------------------------------------------------------------------
template <bool BT>
__global__ __launch_bounds__(256) void wmma_gemm_kernel(
    const bf16* __restrict__ A, int lda, long long strideA,
    const bf16* __restrict__ Bm, int ldb, long long strideB,
    float* __restrict__ C, int ldc, long long strideC,
    const float* __restrict__ bias1, const float* __restrict__ bias2,
    int M, int Ncols, int K)
{
  int batch = blockIdx.z;
  A  += (size_t)batch * strideA;
  Bm += (size_t)batch * strideB;
  C  += (size_t)batch * strideC;

  int wave = threadIdx.x >> 5, lane = threadIdx.x & 31;
  int tiles_n = Ncols >> 6;                 // 64-wide panels
  int tiles_m = M >> 4;
  int tile = blockIdx.x * 8 + wave;
  if (tile >= tiles_m * tiles_n) return;
  int r0 = (tile / tiles_n) * 16;
  int n0 = (tile % tiles_n) * 64;

  f32x8 acc[4];
#pragma unroll
  for (int nt = 0; nt < 4; ++nt) acc[nt] = (f32x8){};

  const bf16* Arow = A + (size_t)r0 * lda;
  for (int k0 = 0; k0 < K; k0 += 32) {
    bf16x16 a = load_a_tile(Arow, lda, k0, lane);
    if (k0 + 32 < K)   // warm next A K-panel (global_prefetch_b8)
      __builtin_prefetch(Arow + (size_t)(lane & 15) * lda + k0 + 32, 0, 0);
#pragma unroll
    for (int nt = 0; nt < 4; ++nt) {
      bf16x16 b = BT ? load_b_trans(Bm, ldb, n0 + nt * 16, k0, lane)
                     : load_b_norm(Bm, ldb, n0 + nt * 16, k0, lane);
      acc[nt] = __builtin_amdgcn_wmma_f32_16x16x32_bf16(false, a, false, b,
                                                        (short)0, acc[nt],
                                                        false, false);
    }
  }

  int c = lane & 15, h = lane >> 4;
#pragma unroll
  for (int nt = 0; nt < 4; ++nt) {
    int col = n0 + nt * 16 + c;
    float bv = (bias1 ? bias1[col] : 0.0f) + (bias2 ? bias2[col] : 0.0f);
#pragma unroll
    for (int e = 0; e < 8; ++e) {
      int row = r0 + h * 8 + e;
      C[(size_t)row * ldc + col] = acc[nt][e] + bv;
    }
  }
}

// ---------------------------------------------------------------------------
// fp32 -> bf16 conversion (vectorized by 4)
// ---------------------------------------------------------------------------
__global__ void cvt_f32_bf16_kernel(const float* __restrict__ in,
                                    bf16* __restrict__ out, int n4) {
  int i = blockIdx.x * blockDim.x + threadIdx.x;
  if (i >= n4) return;
  float4 v = reinterpret_cast<const float4*>(in)[i];
  bf16x4 o;
  o[0] = (bf16)v.x; o[1] = (bf16)v.y; o[2] = (bf16)v.z; o[3] = (bf16)v.w;
  reinterpret_cast<bf16x4*>(out)[i] = o;
}

// ---------------------------------------------------------------------------
// Attention scores: scores[b,i,j] = sigmoid(sum_n AttW[n]*tanh(Px[b,j,n]+Wx[b,i,n]) + Attb)
// One thread per (b,i,j) triple; N=64 reduced with float4 vectors.
// ---------------------------------------------------------------------------
__global__ void attn_scores_kernel(const float* __restrict__ Wx,
                                   const float* __restrict__ Px,
                                   const float* __restrict__ AttW,
                                   const float* __restrict__ Attb,
                                   bf16* __restrict__ scores) {
  int idx = blockIdx.x * blockDim.x + threadIdx.x;
  if (idx >= B * S * S) return;
  int j = idx & (S - 1);
  int i = (idx >> 7) & (S - 1);
  int b = idx >> 14;
  const float4* px = reinterpret_cast<const float4*>(Px + ((size_t)b * S + j) * N);
  const float4* wx = reinterpret_cast<const float4*>(Wx + ((size_t)b * S + i) * N);
  const float4* aw = reinterpret_cast<const float4*>(AttW);
  float acc = Attb[0];
#pragma unroll
  for (int n = 0; n < N / 4; ++n) {
    float4 p = px[n], w = wx[n], a = aw[n];
    acc += a.x * tanhf(p.x + w.x) + a.y * tanhf(p.y + w.y) +
           a.z * tanhf(p.z + w.z) + a.w * tanhf(p.w + w.w);
  }
  scores[idx] = (bf16)sigmoidf_(acc);
}

// ---------------------------------------------------------------------------
// Sequential LSTM: one 256-thread block per direction/job.
//   gates[8,512] = GX[b,s,:] + h @ Whh^T     (WMMA, h padded to 16 rows)
//   c = f*c + i*g ; h = o*tanh(c)            (elementwise, LDS state)
// Whh (bf16, [512,128] row-major) is preloaded into registers: 16 x v16bf
// per wave (each wave owns 64 gate columns), reused across all 128 steps.
// ---------------------------------------------------------------------------
__global__ __launch_bounds__(256) void lstm_seq_kernel(
    const float* __restrict__ GX0, const bf16* __restrict__ W0, int off0, int rev0,
    const float* __restrict__ GX1, const bf16* __restrict__ W1, int off1, int rev1,
    float* __restrict__ out, int outStride)
{
  const float* GX = (blockIdx.x == 0) ? GX0 : GX1;
  const bf16*  W  = (blockIdx.x == 0) ? W0  : W1;
  int off = (blockIdx.x == 0) ? off0 : off1;
  int rev = (blockIdx.x == 0) ? rev0 : rev1;

  __shared__ bf16  hbf[16 * H];    // h state, bf16, rows 8..15 stay zero
  __shared__ float gbuf[B * G4];   // gate pre-activations
  __shared__ float cst[B * H];     // cell state

  int tid = threadIdx.x;
  int wave = tid >> 5, lane = tid & 31;

  for (int i = tid; i < 16 * H; i += 256) hbf[i] = (bf16)0.0f;
  for (int i = tid; i < B * H;  i += 256) cst[i] = 0.0f;
  __syncthreads();

  // Preload this wave's Whh^T tiles: cols [wave*64, wave*64+63], K = 0..127.
  bf16x16 Breg[16];
#pragma unroll
  for (int cw = 0; cw < 4; ++cw)
#pragma unroll
    for (int kt = 0; kt < 4; ++kt)
      Breg[cw * 4 + kt] = load_b_trans(W, H, wave * 64 + cw * 16, kt * 32, lane);

  for (int t = 0; t < S; ++t) {
    int s = rev ? (S - 1 - t) : t;

    // ---- gate matmul: 4 column tiles x 4 K tiles per wave ----
#pragma unroll
    for (int cw = 0; cw < 4; ++cw) {
      f32x8 acc = {};
#pragma unroll
      for (int kt = 0; kt < 4; ++kt) {
        bf16x16 a = load_a_tile(hbf, H, kt * 32, lane);
        acc = __builtin_amdgcn_wmma_f32_16x16x32_bf16(false, a, false,
                                                      Breg[cw * 4 + kt],
                                                      (short)0, acc, false, false);
      }
      if (lane < 16) {                       // rows 0..7 (real batch rows)
        int col = wave * 64 + cw * 16 + lane;
#pragma unroll
        for (int e = 0; e < 8; ++e) gbuf[e * G4 + col] = acc[e];
      }
    }
    __syncthreads();

    // ---- elementwise LSTM cell over (b, j): 1024 units, 4 per thread ----
#pragma unroll
    for (int it = 0; it < 4; ++it) {
      int u = tid + it * 256;
      int b = u >> 7, j = u & (H - 1);
      const float* gx = GX + (size_t)(b * S + s) * G4;
      float gi = sigmoidf_(gbuf[b * G4 +           j] + gx[          j]);
      float gf = sigmoidf_(gbuf[b * G4 +     H  +  j] + gx[    H  +  j]);
      float gg = tanhf   (gbuf[b * G4 + 2 * H  +  j] + gx[2 * H  +  j]);
      float go = sigmoidf_(gbuf[b * G4 + 3 * H  +  j] + gx[3 * H  +  j]);
      float cc = gf * cst[u] + gi * gg;
      cst[u] = cc;
      float hh = go * tanhf(cc);
      hbf[u] = (bf16)hh;                     // row b, col j (u = b*128+j)
      out[(size_t)(b * S + s) * outStride + off + j] = hh;
      // warm next timestep's gate-input row (global_prefetch_b8)
      if (t + 1 < S) {
        int sn = rev ? (S - 2 - t) : (t + 1);
        __builtin_prefetch(GX + (size_t)(b * S + sn) * G4 + j, 0, 0);
      }
    }
    __syncthreads();
  }
}

// ---------------------------------------------------------------------------
// MLP head: z = h2[:, -1] @ dense_W^T + dense_b ; out = z @ clf_W^T + clf_b
// ---------------------------------------------------------------------------
__global__ __launch_bounds__(64) void head_kernel(
    const float* __restrict__ h2all,  // [B,S,H]
    const float* __restrict__ dW, const float* __restrict__ db,
    const float* __restrict__ cW, const float* __restrict__ cb,
    float* __restrict__ out) {
  __shared__ float z[B * 32];
  int tid = threadIdx.x;
  for (int u = tid; u < B * 32; u += 64) {
    int b = u >> 5, k = u & 31;
    const float* hv = h2all + ((size_t)b * S + (S - 1)) * H;
    const float* w  = dW + (size_t)k * H;
    float acc = db[k];
#pragma unroll 4
    for (int n = 0; n < H; ++n) acc += hv[n] * w[n];
    z[u] = acc;
  }
  __syncthreads();
  if (tid < B * 2) {
    int b = tid >> 1, o = tid & 1;
    const float* w = cW + o * 32;
    float acc = cb[o];
#pragma unroll
    for (int n = 0; n < 32; ++n) acc += z[b * 32 + n] * w[n];
    out[b * 2 + o] = acc;
  }
}

// ---------------------------------------------------------------------------
// Host orchestration
// ---------------------------------------------------------------------------
static inline void cvt(const float* in, bf16* out, int n, hipStream_t stream) {
  int n4 = n / 4;
  cvt_f32_bf16_kernel<<<(n4 + 255) / 256, 256, 0, stream>>>(in, out, n4);
}

extern "C" void kernel_launch(void* const* d_in, const int* in_sizes, int n_in,
                              void* d_out, int out_size, void* d_ws, size_t ws_size,
                              hipStream_t stream) {
  (void)in_sizes; (void)n_in; (void)out_size; (void)ws_size;

  const float* A      = (const float*)d_in[0];
  const float* Wx_W   = (const float*)d_in[1];
  const float* Wx_b   = (const float*)d_in[2];
  const float* Wxh_W  = (const float*)d_in[3];
  const float* Wxh_b  = (const float*)d_in[4];
  const float* Att_W  = (const float*)d_in[5];
  const float* Att_b  = (const float*)d_in[6];
  const float* Wih_f  = (const float*)d_in[7];
  const float* Whh_f  = (const float*)d_in[8];
  const float* bih_f  = (const float*)d_in[9];
  const float* bhh_f  = (const float*)d_in[10];
  const float* Wih_r  = (const float*)d_in[11];
  const float* Whh_r  = (const float*)d_in[12];
  const float* bih_r  = (const float*)d_in[13];
  const float* bhh_r  = (const float*)d_in[14];
  const float* Wih2   = (const float*)d_in[15];
  const float* Whh2   = (const float*)d_in[16];
  const float* bih2   = (const float*)d_in[17];
  const float* bhh2   = (const float*)d_in[18];
  const float* denseW = (const float*)d_in[19];
  const float* denseb = (const float*)d_in[20];
  const float* clfW   = (const float*)d_in[21];
  const float* clfb   = (const float*)d_in[22];
  float* out = (float*)d_out;

  // ---- workspace carve-up (256B aligned) ----
  char* ws = (char*)d_ws;
  size_t off = 0;
  auto take = [&](size_t bytes) -> char* {
    char* p = ws + off;
    off += (bytes + 255) & ~size_t(255);
    return p;
  };
  bf16*  A_bf     = (bf16*)take((size_t)BS * D * 2);
  bf16*  WxW_bf   = (bf16*)take((size_t)N * D * 2);
  bf16*  WxhW_bf  = (bf16*)take((size_t)N * D * 2);
  bf16*  Wihf_bf  = (bf16*)take((size_t)G4 * D * 2);
  bf16*  Wihr_bf  = (bf16*)take((size_t)G4 * D * 2);
  bf16*  Wih2_bf  = (bf16*)take((size_t)G4 * (2 * H) * 2);
  bf16*  Whhf_bf  = (bf16*)take((size_t)G4 * H * 2);
  bf16*  Whhr_bf  = (bf16*)take((size_t)G4 * H * 2);
  bf16*  Whh2_bf  = (bf16*)take((size_t)G4 * H * 2);
  float* WxO      = (float*)take((size_t)BS * N * 4);
  float* PxO      = (float*)take((size_t)BS * N * 4);
  bf16*  sc_bf    = (bf16*)take((size_t)B * S * S * 2);
  float* X        = (float*)take((size_t)BS * D * 4);
  bf16*  X_bf     = (bf16*)take((size_t)BS * D * 2);
  float* GXf      = (float*)take((size_t)BS * G4 * 4);
  float* GXr      = (float*)take((size_t)BS * G4 * 4);
  float* GX2      = (float*)take((size_t)BS * G4 * 4);
  float* Xb       = (float*)take((size_t)BS * 2 * H * 4);
  bf16*  Xb_bf    = (bf16*)take((size_t)BS * 2 * H * 2);
  float* h2all    = (float*)take((size_t)BS * H * 4);

  // ---- bf16 staging of activations + weights ----
  cvt(A,     A_bf,    BS * D,      stream);
  cvt(Wx_W,  WxW_bf,  N * D,       stream);
  cvt(Wxh_W, WxhW_bf, N * D,       stream);
  cvt(Wih_f, Wihf_bf, G4 * D,      stream);
  cvt(Wih_r, Wihr_bf, G4 * D,      stream);
  cvt(Wih2,  Wih2_bf, G4 * 2 * H,  stream);
  cvt(Whh_f, Whhf_bf, G4 * H,      stream);
  cvt(Whh_r, Whhr_bf, G4 * H,      stream);
  cvt(Whh2,  Whh2_bf, G4 * H,      stream);

  // ---- attention projections: Wx / Px  [BS, N] = A_bf @ W^T + b ----
  {
    int tiles = (BS / 16) * (N / 64);                 // 64 panels
    dim3 grid(tiles / 8, 1, 1);
    wmma_gemm_kernel<true><<<grid, 256, 0, stream>>>(
        A_bf, D, 0, WxW_bf, D, 0, WxO, N, 0, Wx_b, nullptr, BS, N, D);
    wmma_gemm_kernel<true><<<grid, 256, 0, stream>>>(
        A_bf, D, 0, WxhW_bf, D, 0, PxO, N, 0, Wxh_b, nullptr, BS, N, D);
  }

  // ---- scores [B,S,S] (bf16) ----
  attn_scores_kernel<<<(B * S * S + 255) / 256, 256, 0, stream>>>(
      WxO, PxO, Att_W, Att_b, sc_bf);

  // ---- X[b] = scores[b] @ A[b]   (batched, B non-transposed) ----
  {
    int tiles = (S / 16) * (D / 64);                  // 64 panels
    dim3 grid(tiles / 8, 1, B);
    wmma_gemm_kernel<false><<<grid, 256, 0, stream>>>(
        sc_bf, S, (long long)S * S,
        A_bf, D, (long long)S * D,
        X, D, (long long)S * D,
        nullptr, nullptr, S, D, S);
  }
  cvt(X, X_bf, BS * D, stream);

  // ---- BiLSTM gate inputs: GXf / GXr [BS, 4H] ----
  {
    int tiles = (BS / 16) * (G4 / 64);                // 512 panels
    dim3 grid(tiles / 8, 1, 1);
    wmma_gemm_kernel<true><<<grid, 256, 0, stream>>>(
        X_bf, D, 0, Wihf_bf, D, 0, GXf, G4, 0, bih_f, bhh_f, BS, G4, D);
    wmma_gemm_kernel<true><<<grid, 256, 0, stream>>>(
        X_bf, D, 0, Wihr_bf, D, 0, GXr, G4, 0, bih_r, bhh_r, BS, G4, D);
  }

  // ---- BiLSTM recurrence: block 0 forward, block 1 reverse -> Xb [BS, 2H] ----
  lstm_seq_kernel<<<2, 256, 0, stream>>>(
      GXf, Whhf_bf, 0,   0,
      GXr, Whhr_bf, H,   1,
      Xb, 2 * H);
  cvt(Xb, Xb_bf, BS * 2 * H, stream);

  // ---- second LSTM gate inputs: GX2 [BS, 4H] = Xb_bf @ Wih2^T + b ----
  {
    int tiles = (BS / 16) * (G4 / 64);
    dim3 grid(tiles / 8, 1, 1);
    wmma_gemm_kernel<true><<<grid, 256, 0, stream>>>(
        Xb_bf, 2 * H, 0, Wih2_bf, 2 * H, 0, GX2, G4, 0, bih2, bhh2, BS, G4, 2 * H);
  }

  // ---- second LSTM recurrence (forward only) -> h2all [BS, H] ----
  lstm_seq_kernel<<<1, 256, 0, stream>>>(
      GX2, Whh2_bf, 0, 0,
      GX2, Whh2_bf, 0, 0,
      h2all, H);

  // ---- MLP head -> out [B, 2] ----
  head_kernel<<<1, 64, 0, stream>>>(h2all, denseW, denseb, clfW, clfb, out);
}
